// SKLayer_83313775607970
// MI455X (gfx1250) — matrix operand order repaired and checked by reading, not compiled, not run on hardware
//
#include <hip/hip_runtime.h>
#include <hip/hip_bf16.h>

typedef __attribute__((ext_vector_type(16))) _Float16 v16h;
typedef __attribute__((ext_vector_type(8)))  float    v8f;

#define BB 4
#define CC 256
#define TT 16
#define NN 1024          // h*w
#define CSTRIDE (TT*NN)  // stride between channels in floats

#define ROWS_PER_WG 128  // row tile (n dimension of aff rows)
#define COLS_PER_CB 128  // column tile per iteration
#define LDA 264          // padded K stride in halves (256 + 8 pad)
#define INV_T (1.0f/0.07f)

#define A_HALFS (ROWS_PER_WG*LDA)          // 128*264
#define B_HALFS (COLS_PER_CB*LDA)          // 128*264
#define SMEM_BYTES (A_HALFS*2 + B_HALFS*2 + 128*4 + 128*4)

union Frag {
    v16h h;
    int4 i[2];
};

// ---------------------------------------------------------------------------
// Kernel 1: per (b,t) slice, fused  aff = S1^T S2  ->  sigmoid -> softmax(/T)
// -> diagonal -> d[b,t,n] = sigmoid(sk_diag) - 0.5.  WMMA f16 16x16x32,
// 2x4 register blocking per wave (8 accumulators, 1.5 ds_b128 per WMMA).
// grid.x = 64 slices * 8 row tiles = 512 blocks, 256 threads (8 wave32).
// ---------------------------------------------------------------------------
__global__ __launch_bounds__(256)
void sk_affinity_diag_kernel(const float* __restrict__ A, float* __restrict__ dbuf) {
    extern __shared__ char smem[];
    _Float16* ldsA   = (_Float16*)smem;                         // [128][LDA]
    _Float16* ldsB   = (_Float16*)(smem + A_HALFS * 2);          // [128][LDA]
    float*    rowsum = (float*)(smem + (A_HALFS + B_HALFS) * 2); // [128]
    float*    diag   = rowsum + 128;                             // [128]

    const int tid   = threadIdx.x;
    const int gid   = blockIdx.x;
    const int slice = gid >> 3;          // b*16 + t
    const int rtile = gid & 7;
    const int b     = slice >> 4;
    const int t     = slice & 15;
    const int t1    = (t > 0)      ? t - 1 : 0;       // seq1 shift
    const int t2    = (t < TT - 1) ? t + 1 : TT - 1;  // seq2 shift
    const int row0  = rtile * ROWS_PER_WG;

    const float* S1 = A + (size_t)b * CC * CSTRIDE + (size_t)t1 * NN; // [c][n]
    const float* S2 = A + (size_t)b * CC * CSTRIDE + (size_t)t2 * NN;

    if (tid < 128) { rowsum[tid] = 0.0f; diag[tid] = 0.0f; }

    // ---- stage A-tile: rows [row0, row0+128) x K=256 channels, f32 -> f16,
    //      stored transposed [n_local][c] so fragments are K-contiguous.
    #pragma unroll
    for (int it = 0; it < 32; ++it) {
        int lin = it * 256 + tid;        // 8192 float4s total
        int c   = lin >> 5;
        int n4  = (lin & 31) << 2;
        float4 v = *(const float4*)(S1 + (size_t)c * CSTRIDE + row0 + n4);
        ldsA[(n4 + 0) * LDA + c] = (_Float16)v.x;
        ldsA[(n4 + 1) * LDA + c] = (_Float16)v.y;
        ldsA[(n4 + 2) * LDA + c] = (_Float16)v.z;
        ldsA[(n4 + 3) * LDA + c] = (_Float16)v.w;
    }

    const int lane  = tid & 31;
    const int wave  = tid >> 5;
    const int rg    = wave & 3;          // row group: 2 fragments (32 rows)
    const int cg    = wave >> 2;         // col group: 4 fragments (64 cols)
    const int lrow  = lane & 15;
    const int khalf = lane >> 4;         // 0: lanes 0-15, 1: lanes 16-31

    for (int cb = 0; cb < 8; ++cb) {
        // ---- stage B col-tile: cols [cb*128, +128) x K=256, transposed [m][c]
        #pragma unroll
        for (int it = 0; it < 32; ++it) {
            int lin = it * 256 + tid;    // 8192 float4s total
            int c   = lin >> 5;
            int m4  = (lin & 31) << 2;
            float4 v = *(const float4*)(S2 + (size_t)c * CSTRIDE + cb * COLS_PER_CB + m4);
            ldsB[(m4 + 0) * LDA + c] = (_Float16)v.x;
            ldsB[(m4 + 1) * LDA + c] = (_Float16)v.y;
            ldsB[(m4 + 2) * LDA + c] = (_Float16)v.z;
            ldsB[(m4 + 3) * LDA + c] = (_Float16)v.w;
        }
        // prefetch next col-tile into cache hierarchy (global_prefetch_b8)
        if (cb < 7)
            __builtin_prefetch(S2 + (size_t)(tid >> 5) * CSTRIDE + (cb + 1) * COLS_PER_CB + (tid & 31) * 4, 0, 1);
        __syncthreads();

        // ---- WMMA: 2x4 tiles per wave, fragments batched per K-step
        v8f acc[2][4];
        #pragma unroll
        for (int i = 0; i < 2; ++i)
            #pragma unroll
            for (int j = 0; j < 4; ++j) acc[i][j] = v8f{};

        #pragma unroll
        for (int kb = 0; kb < 8; ++kb) {
            Frag a[2];
            #pragma unroll
            for (int i = 0; i < 2; ++i) {
                const _Float16* ap = ldsA + ((rg * 2 + i) * 16 + lrow) * LDA + kb * 32 + khalf * 8;
                a[i].i[0] = *(const int4*)ap;         // K = kb*32 + khalf*8 + 0..7
                a[i].i[1] = *(const int4*)(ap + 16);  // K = kb*32 + 16 + khalf*8 + 0..7
            }
            Frag bf[4];
            #pragma unroll
            for (int j = 0; j < 4; ++j) {
                const _Float16* bp = ldsB + ((cg * 4 + j) * 16 + lrow) * LDA + kb * 32 + khalf * 8;
                bf[j].i[0] = *(const int4*)bp;        // K = kb*32 + khalf*8 + 0..7
                bf[j].i[1] = *(const int4*)(bp + 16); // K = kb*32 + 16 + khalf*8 + 0..7
            }
            #pragma unroll
            for (int i = 0; i < 2; ++i)
                #pragma unroll
                for (int j = 0; j < 4; ++j)
                    acc[i][j] = __builtin_amdgcn_wmma_f32_16x16x32_f16(
                        false, a[i].h, false, bf[j].h, (short)0, acc[i][j], false, false);
        }

        // ---- fused epilogue: exp((sigmoid(v)-0.5)/T), row sums + diagonal
        #pragma unroll
        for (int i = 0; i < 2; ++i) {
            #pragma unroll
            for (int j = 0; j < 4; ++j) {
                int gcol = cb * COLS_PER_CB + (cg * 4 + j) * 16 + lrow;  // N = lane&15
                #pragma unroll
                for (int r = 0; r < 8; ++r) {
                    float v  = acc[i][j][r];
                    float sg = 0.5f * tanhf(0.5f * v);            // sigmoid(v)-0.5
                    float p  = __expf(sg * INV_T);
                    int lm   = (rg * 2 + i) * 16 + r + khalf * 8; // local row 0..127
                    if (row0 + lm == gcol) diag[lm] = p;          // exactly one writer
                    float s = p;                                  // reduce over 16 cols
                    s += __shfl_xor(s, 1, 16);
                    s += __shfl_xor(s, 2, 16);
                    s += __shfl_xor(s, 4, 16);
                    s += __shfl_xor(s, 8, 16);
                    if (lrow == 0) atomicAdd(&rowsum[lm], s);     // ds_add_f32
                }
            }
        }
        __syncthreads();
    }

    if (tid < 128) {
        float skd  = diag[tid] / rowsum[tid];                 // softmax diagonal
        float dval = 0.5f * tanhf(0.5f * skd);                // sigmoid(skd)-0.5
        dbuf[(size_t)slice * NN + row0 + tid] = dval;
    }
}

// ---------------------------------------------------------------------------
// Kernel 2: out = leakyrelu(x + seq1 * d), float4 elementwise.
// ---------------------------------------------------------------------------
__global__ __launch_bounds__(256)
void sk_apply_kernel(const float* __restrict__ A, const float* __restrict__ dbuf,
                     float* __restrict__ out) {
    size_t i4 = (size_t)blockIdx.x * 256 + threadIdx.x;
    size_t l  = i4 * 4;                    // linear index, n%4 == 0
    int n  = (int)(l & (NN - 1));
    int t  = (int)((l >> 10) & 15);
    int b  = (int)(l >> 22);
    size_t l1 = (t > 0) ? l - NN : l;      // seq1: shift t back by one

    float4 x  = *(const float4*)(A + l);
    float4 s1 = *(const float4*)(A + l1);
    float4 d  = *(const float4*)(dbuf + ((size_t)(b * 16 + t) << 10) + n);

    float4 o;
    o.x = fmaf(s1.x, d.x, x.x);
    o.y = fmaf(s1.y, d.y, x.y);
    o.z = fmaf(s1.z, d.z, x.z);
    o.w = fmaf(s1.w, d.w, x.w);
    o.x = (o.x >= 0.0f) ? o.x : 0.01f * o.x;
    o.y = (o.y >= 0.0f) ? o.y : 0.01f * o.y;
    o.z = (o.z >= 0.0f) ? o.z : 0.01f * o.z;
    o.w = (o.w >= 0.0f) ? o.w : 0.01f * o.w;
    *(float4*)(out + l) = o;
}

extern "C" void kernel_launch(void* const* d_in, const int* in_sizes, int n_in,
                              void* d_out, int out_size, void* d_ws, size_t ws_size,
                              hipStream_t stream) {
    const float* A    = (const float*)d_in[0];
    float*       out  = (float*)d_out;
    float*       dbuf = (float*)d_ws;   // 64*1024 floats = 256 KB

    // 64 slices * 8 row-tiles = 512 WGs, 256 threads (8 wave32), ~136 KB LDS
    sk_affinity_diag_kernel<<<BB * TT * (NN / ROWS_PER_WG), 256, SMEM_BYTES, stream>>>(A, dbuf);

    // 16.7M elements / 4 per thread / 256 threads = 16384 blocks
    sk_apply_kernel<<<(BB * CC * TT * NN) / 4 / 256, 256, 0, stream>>>(A, dbuf, out);
}
